// MultiHeadSelfAttention_14173392077332
// MI455X (gfx1250) — matrix operand order repaired and checked by reading, not compiled, not run on hardware
//
#include <hip/hip_runtime.h>

// MHSA with IoU-reweighted softmax for gfx1250 (MI455X), wave32 WMMA f16.
// T=1024, B=8, D=1024, H=16, hd=64 (fixed by the reference setup).
//
// Layouts chosen so every WMMA fragment load is per-lane contiguous
// (lowering to global_load_b128 / ds_load_b128):
//   q : (B,H,T,hd)  row-major  -> A-frag contiguous
//   k : (B,H,T,hd)  row-major  -> scores B-frag via frag_b_transposed(ld=64)
//   v : (B,H,hd,T)  transposed -> AV     B-frag via frag_b_transposed(ld=T)
// GEMMs use 2x2 register tiling per wave (32x32 out): 4 WMMAs per
// 2 A + 2 B fragment loads -> 2 b128 loads per WMMA.

typedef __attribute__((ext_vector_type(16))) _Float16 v16h;
typedef __attribute__((ext_vector_type(8)))  float    v8f;

#define T_DIM 1024
#define B_DIM 8
#define D_DIM 1024
#define H_DIM 16
#define HD    64

#define WMMA_F16(A, B, C) \
  __builtin_amdgcn_wmma_f32_16x16x32_f16(false, (A), false, (B), (short)0, (C), false, false)

// ---------------- WMMA fragment loaders (contiguous k = 16*(lane/16) + e, wave32) -------

// A-matrix 16x32 f16, element (m,k) at src[m*ld + k].
__device__ inline v16h frag_a_rowmajor(const _Float16* src, int ld) {
  v16h a;
  const int l   = threadIdx.x & 31;
  const int row = l & 15;
  const int kb  = (l >> 4) << 4;          // 0 or 16
#pragma unroll
  for (int e = 0; e < 16; ++e) {
    const int k = kb + e;
    a[e] = src[row * ld + k];
  }
  return a;
}

// B-matrix 32x16 where memory holds element (n,k) at src[n*ld + k].
// Per-lane contiguous 16 f16 -> vectorizes to b128 loads.
__device__ inline v16h frag_b_transposed(const _Float16* src, int ld) {
  v16h b;
  const int l  = threadIdx.x & 31;
  const int n  = l & 15;
  const int kb = (l >> 4) << 4;           // 0 or 16
#pragma unroll
  for (int e = 0; e < 16; ++e) b[e] = src[n * ld + kb + e];
  return b;
}

// ---------------- Stage 0: fp32 -> f16 conversion ---------------------------

__global__ void cvt_f32_to_f16(const float* __restrict__ in,
                               _Float16* __restrict__ out, int n) {
  int i = blockIdx.x * blockDim.x + threadIdx.x;
  if (i < n) out[i] = (_Float16)in[i];
}

// ---------------- Stage 1: QKV GEMM + head scatter ---------------------------
// qkv[m,n] = sum_k query[m,k]*W_qkv[n,k] + b_qkv[n];  m = t*B + b (row-major).
// q scaled by hd^-0.5.  Block = 64x64 tile, 4 waves in 2x2, wave = 32x32.

__global__ void __launch_bounds__(128)
qkv_gemm(const _Float16* __restrict__ qf16,    // (T*B, D) f16
         const _Float16* __restrict__ w,       // (3D, D)  f16
         const float*    __restrict__ bias,    // (3D)
         _Float16* __restrict__ q_h,           // (B,H,T,hd) scaled
         _Float16* __restrict__ k_h,           // (B,H,T,hd)
         _Float16* __restrict__ vT) {          // (B,H,hd,T)
  const int wv = threadIdx.x >> 5;
  const int m0 = blockIdx.x * 64 + (wv >> 1) * 32;
  const int n0 = blockIdx.y * 64 + (wv & 1) * 32;

  v8f acc[2][2];
#pragma unroll
  for (int i = 0; i < 2; ++i)
#pragma unroll
    for (int j = 0; j < 2; ++j) acc[i][j] = (v8f){};

  for (int k0 = 0; k0 < D_DIM; k0 += 32) {
    v16h a0 = frag_a_rowmajor(qf16 + (size_t)m0 * D_DIM + k0, D_DIM);
    v16h a1 = frag_a_rowmajor(qf16 + (size_t)(m0 + 16) * D_DIM + k0, D_DIM);
    v16h b0 = frag_b_transposed(w + (size_t)n0 * D_DIM + k0, D_DIM);
    v16h b1 = frag_b_transposed(w + (size_t)(n0 + 16) * D_DIM + k0, D_DIM);
    acc[0][0] = WMMA_F16(a0, b0, acc[0][0]);
    acc[0][1] = WMMA_F16(a0, b1, acc[0][1]);
    acc[1][0] = WMMA_F16(a1, b0, acc[1][0]);
    acc[1][1] = WMMA_F16(a1, b1, acc[1][1]);
  }

  const int l  = threadIdx.x & 31;
  const int mb = (l >> 4) << 3;
#pragma unroll
  for (int j = 0; j < 2; ++j) {
    const int n  = n0 + j * 16 + (l & 15);
    const float bn = bias[n];
    const int sec = n >> 10;               // 0=q, 1=k, 2=v
    const int df  = n & (D_DIM - 1);
    const int h   = df >> 6;
    const int d   = df & (HD - 1);
#pragma unroll
    for (int i = 0; i < 2; ++i) {
#pragma unroll
      for (int r = 0; r < 8; ++r) {
        const int m  = m0 + i * 16 + mb + r;
        const int t  = m >> 3;             // B_DIM == 8
        const int bb = m & 7;
        const int bh = bb * H_DIM + h;
        const float val = acc[i][j][r] + bn;
        if (sec == 0)
          q_h[((size_t)bh * T_DIM + t) * HD + d] = (_Float16)(val * 0.125f);
        else if (sec == 1)
          k_h[((size_t)bh * T_DIM + t) * HD + d] = (_Float16)val;
        else
          vT[((size_t)bh * HD + d) * T_DIM + t] = (_Float16)val;
      }
    }
  }
}

// ---------------- Stage 2: streaming IoU attention ---------------------------
// Per wave: 16 query rows of one (b,h). Loop keys in chunks of 32:
//   scores = q·k^T (4 WMMA), w = exp(s)*iou, stage f16 w in LDS,
//   ctx += w @ V (4 WMMA). Normalize by running sum at the end.

__global__ void __launch_bounds__(128)
attn_kernel(const _Float16* __restrict__ q_h,
            const _Float16* __restrict__ k_h,   // (B,H,T,hd)
            const _Float16* __restrict__ vT,    // (B,H,hd,T)
            const float*    __restrict__ ious,  // (B,T,T)
            _Float16* __restrict__ attn) {      // (T,B,D) f16
  __shared__ _Float16 pLDS[4][16][32];
  __shared__ float    red[4][16][16];
  __shared__ float    dn[4][16];

  const int wv = threadIdx.x >> 5;
  const int l  = threadIdx.x & 31;
  const int bh = blockIdx.y;              // b*H + h
  const int b  = bh >> 4;
  const int h  = bh & 15;
  const int t0 = blockIdx.x * 64 + wv * 16;
  const int n  = l & 15;
  const int mb = (l >> 4) << 3;

  const _Float16* kbh = k_h + (size_t)bh * T_DIM * HD;   // keys, row-major
  const _Float16* vbh = vT  + (size_t)bh * HD * T_DIM;   // values, (hd,T)

  const v16h qa0 = frag_a_rowmajor(q_h + ((size_t)bh * T_DIM + t0) * HD, HD);
  const v16h qa1 = frag_a_rowmajor(q_h + ((size_t)bh * T_DIM + t0) * HD + 32, HD);

  v8f ctx[4];
#pragma unroll
  for (int nt = 0; nt < 4; ++nt) ctx[nt] = (v8f){};
  float dsum[8];
#pragma unroll
  for (int r = 0; r < 8; ++r) dsum[r] = 0.f;

  for (int s0 = 0; s0 < T_DIM; s0 += 32) {
    // Prefetch next K/V chunk while this one runs through the matrix pipe.
    if (s0 + 32 < T_DIM) {
      __builtin_prefetch((const char*)(kbh + (size_t)(s0 + 32) * HD) + l * 128, 0, 1);
      __builtin_prefetch((const char*)(vbh + (size_t)l * T_DIM + s0 + 32), 0, 1);
      __builtin_prefetch((const char*)(vbh + (size_t)(l + 32) * T_DIM + s0 + 32), 0, 1);
    }
#pragma unroll
    for (int half = 0; half < 2; ++half) {
      const int sbase = s0 + half * 16;
      // B[K=d][N=key] = k[key][d] -> transposed loader, per-lane contiguous
      v16h kb0 = frag_b_transposed(kbh + (size_t)sbase * HD +  0, HD);
      v16h kb1 = frag_b_transposed(kbh + (size_t)sbase * HD + 32, HD);
      v8f sc = {};
      sc = WMMA_F16(qa0, kb0, sc);
      sc = WMMA_F16(qa1, kb1, sc);
#pragma unroll
      for (int r = 0; r < 8; ++r) {
        const int t = t0 + mb + r;
        const int s = sbase + n;
        const float wgt = __expf(sc[r]) * ious[((size_t)b * T_DIM + t) * T_DIM + s];
        dsum[r] += wgt;
        pLDS[wv][mb + r][half * 16 + n] = (_Float16)wgt;
      }
    }
    __syncthreads();
    const v16h pa = frag_a_rowmajor(&pLDS[wv][0][0], 32);
#pragma unroll
    for (int nt = 0; nt < 4; ++nt) {
      // B[K=key][N=d] = v[key][d] = vT[d][key] -> transposed loader
      v16h bv = frag_b_transposed(vbh + (size_t)(nt * 16) * T_DIM + s0, T_DIM);
      ctx[nt] = WMMA_F16(pa, bv, ctx[nt]);
    }
    __syncthreads();
  }

  // Reduce denominators across the 16 lanes of each row group.
#pragma unroll
  for (int r = 0; r < 8; ++r) red[wv][mb + r][n] = dsum[r];
  __syncthreads();
  if (l < 16) {
    float s = 0.f;
#pragma unroll
    for (int j = 0; j < 16; ++j) s += red[wv][l][j];
    dn[wv][l] = s;
  }
  __syncthreads();

  // Normalize and scatter to (T,B,D) f16.
#pragma unroll
  for (int r = 0; r < 8; ++r) {
    const int row = mb + r;
    const float inv = 1.0f / dn[wv][row];
    const int t = t0 + row;
    const size_t base = ((size_t)t * B_DIM + b) * D_DIM + h * HD;
#pragma unroll
    for (int nt = 0; nt < 4; ++nt)
      attn[base + nt * 16 + n] = (_Float16)(ctx[nt][r] * inv);
  }
}

// ---------------- Stage 3: output projection --------------------------------

__global__ void __launch_bounds__(128)
out_gemm(const _Float16* __restrict__ x,     // (T*B, D) f16
         const _Float16* __restrict__ w,     // (D, D)   f16
         const float*    __restrict__ bias,  // (D)
         float* __restrict__ out) {          // (T,B,D) fp32
  const int wv = threadIdx.x >> 5;
  const int m0 = blockIdx.x * 64 + (wv >> 1) * 32;
  const int n0 = blockIdx.y * 64 + (wv & 1) * 32;

  v8f acc[2][2];
#pragma unroll
  for (int i = 0; i < 2; ++i)
#pragma unroll
    for (int j = 0; j < 2; ++j) acc[i][j] = (v8f){};

  for (int k0 = 0; k0 < D_DIM; k0 += 32) {
    v16h a0 = frag_a_rowmajor(x + (size_t)m0 * D_DIM + k0, D_DIM);
    v16h a1 = frag_a_rowmajor(x + (size_t)(m0 + 16) * D_DIM + k0, D_DIM);
    v16h b0 = frag_b_transposed(w + (size_t)n0 * D_DIM + k0, D_DIM);
    v16h b1 = frag_b_transposed(w + (size_t)(n0 + 16) * D_DIM + k0, D_DIM);
    acc[0][0] = WMMA_F16(a0, b0, acc[0][0]);
    acc[0][1] = WMMA_F16(a0, b1, acc[0][1]);
    acc[1][0] = WMMA_F16(a1, b0, acc[1][0]);
    acc[1][1] = WMMA_F16(a1, b1, acc[1][1]);
  }

  const int l  = threadIdx.x & 31;
  const int mb = (l >> 4) << 3;
#pragma unroll
  for (int j = 0; j < 2; ++j) {
    const int n  = n0 + j * 16 + (l & 15);
    const float bn = bias[n];
#pragma unroll
    for (int i = 0; i < 2; ++i)
#pragma unroll
      for (int r = 0; r < 8; ++r)
        out[(size_t)(m0 + i * 16 + mb + r) * D_DIM + n] = acc[i][j][r] + bn;
  }
}

// ---------------- Host-side launcher ----------------------------------------

extern "C" void kernel_launch(void* const* d_in, const int* in_sizes, int n_in,
                              void* d_out, int out_size, void* d_ws, size_t ws_size,
                              hipStream_t stream) {
  const float* query = (const float*)d_in[0];
  const float* ious  = (const float*)d_in[1];
  const float* Wqkv  = (const float*)d_in[2];
  const float* bqkv  = (const float*)d_in[3];
  const float* Wout  = (const float*)d_in[4];
  const float* bout  = (const float*)d_in[5];
  float* out = (float*)d_out;

  const size_t N_TOK  = (size_t)T_DIM * B_DIM * D_DIM;  // 8,388,608
  const size_t N_WQKV = (size_t)3 * D_DIM * D_DIM;      // 3,145,728
  const size_t N_WOUT = (size_t)D_DIM * D_DIM;          // 1,048,576

  char* ws = (char*)d_ws;
  _Float16* q_h    = (_Float16*)ws; ws += N_TOK  * sizeof(_Float16);
  _Float16* k_h    = (_Float16*)ws; ws += N_TOK  * sizeof(_Float16);
  _Float16* vT     = (_Float16*)ws; ws += N_TOK  * sizeof(_Float16);
  _Float16* attn   = (_Float16*)ws; ws += N_TOK  * sizeof(_Float16);
  _Float16* qf16   = (_Float16*)ws; ws += N_TOK  * sizeof(_Float16);
  _Float16* wqkv16 = (_Float16*)ws; ws += N_WQKV * sizeof(_Float16);
  _Float16* wout16 = (_Float16*)ws; ws += N_WOUT * sizeof(_Float16);

  cvt_f32_to_f16<<<dim3((N_TOK  + 255) / 256), 256, 0, stream>>>(query, qf16,  (int)N_TOK);
  cvt_f32_to_f16<<<dim3((N_WQKV + 255) / 256), 256, 0, stream>>>(Wqkv, wqkv16, (int)N_WQKV);
  cvt_f32_to_f16<<<dim3((N_WOUT + 255) / 256), 256, 0, stream>>>(Wout, wout16, (int)N_WOUT);

  // M = T*B = 8192 rows.
  dim3 g1(8192 / 64, 3 * D_DIM / 64);    // (128, 48)
  qkv_gemm<<<g1, 128, 0, stream>>>(qf16, wqkv16, bqkv, q_h, k_h, vT);

  dim3 g2(T_DIM / 64, B_DIM * H_DIM);    // (16, 128)
  attn_kernel<<<g2, 128, 0, stream>>>(q_h, k_h, vT, ious, attn);

  dim3 g3(8192 / 64, D_DIM / 64);        // (128, 16)
  out_gemm<<<g3, 128, 0, stream>>>(attn, wout16, bout, out);
}
